// Seq2Seq_88106959110179
// MI455X (gfx1250) — matrix operand order repaired and checked by reading, not compiled, not run on hardware
//
#include <hip/hip_runtime.h>
#include <hip/hip_bf16.h>
#include <math.h>

// ---------------------------------------------------------------------------
// Problem constants (from the reference)
// ---------------------------------------------------------------------------
#define VSZ   10000
#define ESZ   16
#define HSZ   128
#define G3H   384        // 3*H
#define KCONV 11
#define PADC  5
#define BB    32
#define TENC  2048
#define TDEC  128
#define TOUT  127        // TDEC-1 decoder steps / outputs

typedef _Float16 v16h __attribute__((ext_vector_type(16)));
typedef float    v8f  __attribute__((ext_vector_type(8)));

__device__ __forceinline__ float sigmoidf_(float x) { return 1.0f / (1.0f + expf(-x)); }

__device__ __forceinline__ void cvt8(v16h& f, int base, float4 a, float4 b) {
    f[base + 0] = (_Float16)a.x; f[base + 1] = (_Float16)a.y;
    f[base + 2] = (_Float16)a.z; f[base + 3] = (_Float16)a.w;
    f[base + 4] = (_Float16)b.x; f[base + 5] = (_Float16)b.y;
    f[base + 6] = (_Float16)b.z; f[base + 7] = (_Float16)b.w;
}

// ---------------------------------------------------------------------------
// WMMA GEMM:  C[m,n] = sum_k A[m,k] * W[n,k] + bias[n]
//   A: (M x K) row-major f32; if GATHER, row m is A + a_idx[m]*lda
//      (fuses the embedding lookup into the gi0 GEMM).
//   W: (N x K) row-major f32  (we compute A @ W^T, PyTorch-style).
// One wave computes one 16x16 tile via v_wmma_f32_16x16x32_f16.
// K is compile-time: all bounds predicates fold, K-loop fully unrolls,
// fragment loads become unconditional global_load_b128.
//
// ISA 7.12.2 fragment layouts used here (wave32):
//   A 16x32 f16 : row = lane%16; lanes 0-15 hold K {0..7,16..23},
//                 lanes 16-31 hold K {8..15,24..31}  -> two 8-float runs
//   B 32x16 f16 : col = lane%16; lanes 0-15 hold K {0..7,16..23},
//                 lanes 16-31 hold K {8..15,24..31}  -> two 8-float runs
//   C/D f32     : col = lane%16; acc[r] = row r + 8*(lane>=16)
// ---------------------------------------------------------------------------
template<int K, bool GATHER>
__global__ void gemm_wmma_f16(const float* __restrict__ A,
                              const int*   __restrict__ a_idx,
                              int lda,
                              const float* __restrict__ W, int ldw,
                              const float* __restrict__ bias,
                              float* __restrict__ C, int ldc,
                              int M, int N)
{
    const int wave = threadIdx.x >> 5;
    const int lane = threadIdx.x & 31;
    const int tilesN = N >> 4;
    const int tilesM = M >> 4;
    const int tile = blockIdx.x * (blockDim.x >> 5) + wave;
    if (tile >= tilesM * tilesN) return;

    const int tm  = tile / tilesN;
    const int tn  = tile - tm * tilesN;
    const int l16 = lane & 15;
    const int hi  = lane >> 4;                // lane group 0 / 1

    const int row  = tm * 16 + l16;           // A row handled by this lane
    const int coln = tn * 16 + l16;           // B column handled by this lane

    const float* __restrict__ Arow =
        GATHER ? (A + (size_t)a_idx[row] * lda) : (A + (size_t)row * lda);
    const float* __restrict__ Wrow = W + (size_t)coln * ldw;

    v8f acc = {};
#pragma unroll
    for (int k0 = 0; k0 < K; k0 += 32) {
        v16h af, bf;

        // ---- A fragment: run0 = k0+8*hi .. +7 (always valid: k0 < K)
        {
            const float4* p = (const float4*)(Arow + k0 + 8 * hi);
            cvt8(af, 0, p[0], p[1]);
        }
        // run1 = k0+16+8*hi .. +7 (valid iff k0+16 < K; hi-independent
        // because K and k0+16 are multiples of 16)
        if (k0 + 16 < K) {
            const float4* p = (const float4*)(Arow + k0 + 16 + 8 * hi);
            cvt8(af, 8, p[0], p[1]);
        } else {
#pragma unroll
            for (int i = 8; i < 16; ++i) af[i] = (_Float16)0.0f;
        }

        // ---- B fragment: run0 = k0+8*hi .. +7, run1 = k0+16+8*hi .. +7
        if (k0 + 31 < K) {                    // fully valid for both halves
            const float4* p = (const float4*)(Wrow + k0 + 8 * hi);
            cvt8(bf, 0, p[0], p[1]);
            const float4* p2 = (const float4*)(Wrow + k0 + 16 + 8 * hi);
            cvt8(bf, 8, p2[0], p2[1]);
        } else {                              // K==16 tail: slots 8..15 are 0
            const float4* p = (const float4*)(Wrow + k0 + 8 * hi);
            cvt8(bf, 0, p[0], p[1]);
#pragma unroll
            for (int i = 8; i < 16; ++i) bf[i] = (_Float16)0.0f;
        }

        acc = __builtin_amdgcn_wmma_f32_16x16x32_f16(
                  /*neg_a=*/false, af, /*neg_b=*/false, bf,
                  /*c_mod=*/(short)0, acc, /*reuse_a=*/false, /*reuse_b=*/false);
    }

    const float bv = bias ? bias[coln] : 0.0f;
#pragma unroll
    for (int r = 0; r < 8; ++r) {
        int rr = tm * 16 + r + 8 * hi;
        C[(size_t)rr * ldc + (tn * 16 + l16)] = acc[r] + bv;
    }
}

// ---------------------------------------------------------------------------
// GRU layer recurrence (PyTorch GRUCell math), gi precomputed.
//   gi: (B, T, 3H) with b_ih already added.   ys: (B, T, H)
// One workgroup per batch element; 384 threads (one per gate row).
// h lives in LDS; Whh (384x128, 192KB) streams from L2 each step.
// ---------------------------------------------------------------------------
__global__ void gru_rec_kernel(const float* __restrict__ gi,
                               const float* __restrict__ Whh,
                               const float* __restrict__ bhh,
                               float* __restrict__ ys, int T)
{
    const int b = blockIdx.x;
    const int g = threadIdx.x;             // 0..383
    __shared__ float h[HSZ];
    __shared__ float gh[G3H];

    if (g < HSZ) h[g] = 0.0f;
    __syncthreads();

    const float* __restrict__ wr = Whh + (size_t)g * HSZ;
    const float  bh = bhh[g];

    for (int t = 0; t < T; ++t) {
        const float* __restrict__ girow = gi + ((size_t)b * T + t) * G3H;
        // gh[g] = h . Whh[g] + bhh[g]
        float s = bh;
#pragma unroll 8
        for (int j = 0; j < HSZ; ++j) s += h[j] * wr[j];
        gh[g] = s;
        __syncthreads();

        float hn = 0.0f;
        if (g < HSZ) {
            float r = sigmoidf_(girow[g]         + gh[g]);
            float z = sigmoidf_(girow[HSZ + g]   + gh[HSZ + g]);
            float n = tanhf   (girow[2*HSZ + g]  + r * gh[2*HSZ + g]);
            hn = (1.0f - z) * n + z * h[g];
        }
        __syncthreads();                   // all reads of h/gh done
        if (g < HSZ) {
            h[g] = hn;
            ys[((size_t)b * T + t) * HSZ + g] = hn;
        }
        __syncthreads();                   // h visible for next step
    }
}

// ---------------------------------------------------------------------------
// Decoder: attention (+conv over previous attention weights) + GRU cell,
// 127 sequential steps. One workgroup (256 threads) per batch element.
// dec_out: (B, 127, H) = hx_new + sx
// ---------------------------------------------------------------------------
__global__ void decoder_kernel(const float* __restrict__ eh,      // (B,TENC,H)
                               const int*   __restrict__ y,       // (B,TDEC)
                               const float* __restrict__ emb,     // (V,E)
                               const float* __restrict__ dWih,    // (3H, E+H)
                               const float* __restrict__ dWhh,    // (3H, H)
                               const float* __restrict__ dbih,
                               const float* __restrict__ dbhh,
                               const float* __restrict__ h_init,  // (1,H)
                               const float* __restrict__ conv_w,  // (11)
                               const float* __restrict__ conv_b,  // (1)
                               float* __restrict__ dec_out)
{
    const int b   = blockIdx.x;
    const int tid = threadIdx.x;           // 0..255

    __shared__ float ax [TENC];            // previous attention weights
    __shared__ float pax[TENC];            // scores -> softmax in place
    __shared__ float hx [HSZ];
    __shared__ float sx [HSZ];
    __shared__ float sxp[256];
    __shared__ float gi [G3H];
    __shared__ float gh [G3H];
    __shared__ float ix [ESZ];
    __shared__ float red[256];

    for (int t = tid; t < TENC; t += 256) ax[t] = 0.0f;
    if (tid < HSZ) hx[tid] = h_init[tid];
    __syncthreads();

    const float cb = conv_b[0];
    const float* __restrict__ ehb = eh + (size_t)b * TENC * HSZ;

    for (int step = 0; step < TOUT; ++step) {
        // ---- scores: pax[t] = eh[b,t,:].hx (+ conv(ax) + conv_b if step>0)
        for (int t = tid; t < TENC; t += 256) {
            const float* __restrict__ er = ehb + (size_t)t * HSZ;
            float s = 0.0f;
#pragma unroll 8
            for (int j = 0; j < HSZ; ++j) s += er[j] * hx[j];
            if (step > 0) {
                float c = cb;
#pragma unroll
                for (int k = 0; k < KCONV; ++k) {
                    int tt = t + k - PADC;
                    if (tt >= 0 && tt < TENC) c += conv_w[k] * ax[tt];
                }
                s += c;
            }
            pax[t] = s;
        }
        __syncthreads();

        // ---- softmax over TENC (block reduction)
        float lm = -3.4e38f;
        for (int t = tid; t < TENC; t += 256) lm = fmaxf(lm, pax[t]);
        red[tid] = lm; __syncthreads();
        for (int s = 128; s > 0; s >>= 1) {
            if (tid < s) red[tid] = fmaxf(red[tid], red[tid + s]);
            __syncthreads();
        }
        const float m = red[0]; __syncthreads();

        float ls = 0.0f;
        for (int t = tid; t < TENC; t += 256) {
            float e = expf(pax[t] - m);
            pax[t] = e;                    // store un-normalized weight
            ls += e;
        }
        red[tid] = ls; __syncthreads();
        for (int s = 128; s > 0; s >>= 1) {
            if (tid < s) red[tid] += red[tid + s];
            __syncthreads();
        }
        const float inv = 1.0f / red[0]; __syncthreads();
        for (int t = tid; t < TENC; t += 256) pax[t] *= inv;  // pax == a
        __syncthreads();

        // ---- sx[h] = sum_t a[t] * eh[b,t,h]   (two halves of t-range)
        {
            int h = tid & (HSZ - 1);
            int half = tid >> 7;
            float s = 0.0f;
            for (int t = half * (TENC / 2); t < (half + 1) * (TENC / 2); ++t)
                s += pax[t] * ehb[(size_t)t * HSZ + h];
            sxp[tid] = s;
        }
        __syncthreads();
        if (tid < HSZ) sx[tid] = sxp[tid] + sxp[tid + HSZ];
        // ---- ix = emb[y[b, step]]
        if (tid >= 128 && tid < 128 + ESZ)
            ix[tid - 128] = emb[(size_t)y[b * TDEC + step] * ESZ + (tid - 128)];
        __syncthreads();

        // ---- gi[g] = [ix, sx] . dWih[g] + dbih[g] ; gh[g] = hx . dWhh[g] + dbhh[g]
        for (int g = tid; g < G3H; g += 256) {
            const float* __restrict__ wr = dWih + (size_t)g * (ESZ + HSZ);
            float s = dbih[g];
#pragma unroll
            for (int j = 0; j < ESZ; ++j) s += ix[j] * wr[j];
#pragma unroll 8
            for (int j = 0; j < HSZ; ++j) s += sx[j] * wr[ESZ + j];
            gi[g] = s;

            const float* __restrict__ wh = dWhh + (size_t)g * HSZ;
            float s2 = dbhh[g];
#pragma unroll 8
            for (int j = 0; j < HSZ; ++j) s2 += hx[j] * wh[j];
            gh[g] = s2;
        }
        __syncthreads();

        // ---- GRU cell + output
        if (tid < HSZ) {
            float r = sigmoidf_(gi[tid]          + gh[tid]);
            float z = sigmoidf_(gi[HSZ + tid]    + gh[HSZ + tid]);
            float n = tanhf   (gi[2*HSZ + tid]   + r * gh[2*HSZ + tid]);
            float hn = (1.0f - z) * n + z * hx[tid];
            hx[tid] = hn;                                   // only own slot
            dec_out[((size_t)b * TOUT + step) * HSZ + tid] = hn + sx[tid];
        }
        __syncthreads();

        // ---- carry attention weights to next step
        for (int t = tid; t < TENC; t += 256) ax[t] = pax[t];
        __syncthreads();
    }
}

// ---------------------------------------------------------------------------
// Launch
// ---------------------------------------------------------------------------
extern "C" void kernel_launch(void* const* d_in, const int* in_sizes, int n_in,
                              void* d_out, int out_size, void* d_ws, size_t ws_size,
                              hipStream_t stream)
{
    (void)in_sizes; (void)n_in; (void)out_size; (void)ws_size;

    const int*   x        = (const int*)  d_in[0];   // (B,TENC)
    const int*   y        = (const int*)  d_in[1];   // (B,TDEC)
    const float* emb      = (const float*)d_in[2];   // (V,E)
    const float* enc_Wih0 = (const float*)d_in[3];   // (3H,E)
    const float* enc_Whh0 = (const float*)d_in[4];   // (3H,H)
    const float* enc_bih0 = (const float*)d_in[5];
    const float* enc_bhh0 = (const float*)d_in[6];
    const float* enc_Wih1 = (const float*)d_in[7];   // (3H,H)
    const float* enc_Whh1 = (const float*)d_in[8];
    const float* enc_bih1 = (const float*)d_in[9];
    const float* enc_bhh1 = (const float*)d_in[10];
    const float* dec_Wih  = (const float*)d_in[11];  // (3H,E+H)
    const float* dec_Whh  = (const float*)d_in[12];
    const float* dec_bih  = (const float*)d_in[13];
    const float* dec_bhh  = (const float*)d_in[14];
    const float* h_init   = (const float*)d_in[15];  // (1,H)
    const float* conv_w   = (const float*)d_in[16];  // (1,1,11)
    const float* conv_b   = (const float*)d_in[17];  // (1)
    const float* fc_w     = (const float*)d_in[18];  // (V,H)
    const float* fc_b     = (const float*)d_in[19];  // (V)
    float*       out      = (float*)d_out;           // (B,TOUT,V)

    // Workspace layout (floats): GI | H1 | EH | DEC  (~170 MB total)
    const size_t BT = (size_t)BB * TENC;             // 65536
    float* GI  = (float*)d_ws;                       // (B*TENC, 3H)
    float* H1  = GI  + BT * G3H;                     // (B*TENC, H)
    float* EH  = H1  + BT * HSZ;                     // (B*TENC, H)
    float* DEC = EH  + BT * HSZ;                     // (B*TOUT, H)

    const int GEMM_BLK = 128;                        // 4 waves -> 4 tiles/block
    auto blocks_for = [](long long tiles) { return (int)((tiles + 3) / 4); };

    // 1) gi0 = emb[x] @ enc_Wih0^T + bih0   (gathered A, M=65536, N=384, K=16)
    {
        long long tiles = (long long)(BT / 16) * (G3H / 16);
        gemm_wmma_f16<ESZ, true><<<blocks_for(tiles), GEMM_BLK, 0, stream>>>(
            emb, x, ESZ, enc_Wih0, ESZ, enc_bih0, GI, G3H,
            (int)BT, G3H);
    }
    // 2) encoder layer 0 recurrence -> H1
    gru_rec_kernel<<<BB, G3H, 0, stream>>>(GI, enc_Whh0, enc_bhh0, H1, TENC);

    // 3) gi1 = H1 @ enc_Wih1^T + bih1   (M=65536, N=384, K=128)
    {
        long long tiles = (long long)(BT / 16) * (G3H / 16);
        gemm_wmma_f16<HSZ, false><<<blocks_for(tiles), GEMM_BLK, 0, stream>>>(
            H1, nullptr, HSZ, enc_Wih1, HSZ, enc_bih1, GI, G3H,
            (int)BT, G3H);
    }
    // 4) encoder layer 1 recurrence -> EH
    gru_rec_kernel<<<BB, G3H, 0, stream>>>(GI, enc_Whh1, enc_bhh1, EH, TENC);

    // 5) decoder with attention -> DEC (B, TOUT, H)
    decoder_kernel<<<BB, 256, 0, stream>>>(
        EH, y, emb, dec_Wih, dec_Whh, dec_bih, dec_bhh,
        h_init, conv_w, conv_b, DEC);

    // 6) logits = DEC @ fc_w^T + fc_b   (M=4064, N=10000, K=128)
    {
        long long tiles = (long long)((BB * TOUT) / 16) * (VSZ / 16);
        gemm_wmma_f16<HSZ, false><<<blocks_for(tiles), GEMM_BLK, 0, stream>>>(
            DEC, nullptr, HSZ, fc_w, HSZ, fc_b, out, VSZ,
            BB * TOUT, VSZ);
    }
}
